// _UserModel_19937238188374
// MI455X (gfx1250) — matrix operand order repaired and checked by reading, not compiled, not run on hardware
//
#include <hip/hip_runtime.h>
#include <hip/hip_bf16.h>

typedef _Float16 f16;
typedef __attribute__((ext_vector_type(16))) _Float16 v16h;
typedef __attribute__((ext_vector_type(8)))  float    v8f;

#define DD 64     // embedding dim
#define PP 40     // P == Q == P2
#define RP 48     // rows padded to 3 x 16 WMMA tiles
#define BB 512

// ---------------- WMMA fragment helpers (CDNA5 wave32 layouts) ----------------

// A fragment: 16x32 f16 tile from row-major f16 LDS buffer (ld = row stride)
__device__ __forceinline__ v16h load_a_frag(const f16* A, int ld, int row0, int k0, int lane) {
  const int m  = lane & 15;
  const int hi = (lane >> 4) & 1;
  const f16* r = A + (row0 + m) * ld + k0;
  v16h a;
#pragma unroll
  for (int i = 0; i < 16; ++i) {
    int k = i + 16 * hi;   // contiguous k = 16*hi + i, matching packed B
    a[i] = r[k];
  }
  return a;
}

// B fragment from pre-packed f16 weights: one contiguous 32B load per lane.
// Packed layout: tile t = (nt/16)*kTiles + (kt/32); within tile: lane*16 halves.
__device__ __forceinline__ v16h load_b_packed(const f16* Wp, int kTiles, int ktI, int ntI, int lane) {
  const f16* p = Wp + (((size_t)ntI * kTiles + ktI) << 9) + lane * 16;
  return *(const v16h*)p;
}

// Generic fused GEMM: Dst = act(bn(A[RT*16 x K] @ W[K x N] + bias))
// act: 0 = none, 1 = LeakyReLU(0.2), 2 = ReLU. Optional f16 LDS dest and/or f32 global dest.
template <int RT>
__device__ __forceinline__ void wgemm(const f16* A, int aLd, int K,
                                      const f16* Wp, int N, const float* bias,
                                      const float* bng, const float* bnb,
                                      const float* bnm, const float* bnv,
                                      int act,
                                      f16* Dst, int dLd, int dColOff,
                                      float* GDst, int gLd, int lane) {
  const int kTiles = K >> 5;
  const int n  = lane & 15;
  const int hi = (lane >> 4) & 1;
  for (int nt = 0; nt < N; nt += 16) {
    v8f acc[RT];
#pragma unroll
    for (int i = 0; i < RT; ++i) { v8f z = {}; acc[i] = z; }
    for (int ktI = 0; ktI < kTiles; ++ktI) {
      v16h bf = load_b_packed(Wp, kTiles, ktI, nt >> 4, lane);
#pragma unroll
      for (int i = 0; i < RT; ++i) {
        v16h af = load_a_frag(A, aLd, i * 16, ktI * 32, lane);
        acc[i] = __builtin_amdgcn_wmma_f32_16x16x32_f16(
            /*neg_a=*/false, af, /*neg_b=*/false, bf,
            /*c_mod=*/(short)0, acc[i], /*reuse_a=*/false, /*reuse_b=*/false);
      }
    }
    const int col = nt + n;
    const float bb = bias ? bias[col] : 0.0f;
    float sc = 1.0f, mu = 0.0f, ga = 1.0f, be = 0.0f;
    if (bng) { sc = rsqrtf(bnv[col] + 1e-5f); mu = bnm[col]; ga = bng[col]; be = bnb[col]; }
#pragma unroll
    for (int i = 0; i < RT; ++i) {
#pragma unroll
      for (int r = 0; r < 8; ++r) {
        float v = acc[i][r] + bb;
        if (bng) v = (v - mu) * sc * ga + be;
        if (act == 1) v = (v > 0.0f) ? v : 0.2f * v;
        if (act == 2) v = fmaxf(v, 0.0f);
        const int row = i * 16 + r + 8 * hi;   // ISA f32 C/D 16x16 layout
        if (Dst)  Dst[row * dLd + dColOff + col] = (f16)v;
        if (GDst) GDst[(size_t)row * gLd + col] = v;
      }
    }
  }
}

// Small dense matvec: y[c] = act(sum_k x[k]*W[k*N+c] + b[c]); lanes split columns.
__device__ __forceinline__ void dense_vec(const float* x, const float* W, const float* b,
                                          float* y, int K, int N, int lane, int act) {
  for (int c = lane; c < N; c += 32) {
    float s = b[c];
    for (int k = 0; k < K; ++k) s += x[k] * W[(size_t)k * N + c];
    if (act == 2) s = fmaxf(s, 0.0f);
    y[c] = s;
  }
}

// ---------------- Prep kernel: pack f32 weight W[K][N] into WMMA B-fragment f16 order ----------
// grid = (K/32)*(N/16) tiles, 32 lanes. Output tile: 512 halves; per-lane 16 contiguous.
__global__ __launch_bounds__(32) void k_pack_w(const float* __restrict__ W, int K, int N,
                                               f16* __restrict__ dst) {
  const int t = blockIdx.x;
  const int kTiles = K >> 5;
  const int ntIdx = t / kTiles;
  const int ktIdx = t - ntIdx * kTiles;
  const int lane = threadIdx.x;
  const int col  = (ntIdx << 4) + (lane & 15);
  const int krow = (ktIdx << 5) + ((lane >> 4) << 4);   // K = kt + 16*hi + i
  const float* src = W + (size_t)krow * N + col;
  v16h v;
#pragma unroll
  for (int i = 0; i < 16; ++i) v[i] = (f16)src[(size_t)i * N];
  *(v16h*)(dst + ((size_t)t << 9) + lane * 16) = v;
}

// ---------------- Kernel 1/2: fused embed -> gv-MLP -> attention -> softmax -> agg ----------------
// One wave per group g (first order: g = b, G = B; second order: g = b*Q + q, G = B*Q)
__global__ __launch_bounds__(32) void k_attn_agg(
    const int* __restrict__ pairs,      // [G, PP, 2] (item_id, rate_id)
    const int* __restrict__ users,      // [G] owning user id
    const float* __restrict__ item_tab, const float* __restrict__ rate_tab,
    const float* __restrict__ user_tab,
    const f16* __restrict__ gvW1p, const float* __restrict__ gvb1,
    const f16* __restrict__ gvW2p, const float* __restrict__ gvb2,
    const f16* __restrict__ wWp,   const float* __restrict__ wW,
    const float* __restrict__ wb,
    const f16* __restrict__ aW1p,  const float* __restrict__ ab1,
    const float* __restrict__ aW2,  const float* __restrict__ ab2,
    const float* __restrict__ agW,  const float* __restrict__ agb,
    float* __restrict__ out)            // [G, DD]
{
  const int g = blockIdx.x;
  const int lane = threadIdx.x;
  __shared__ f16  sX [RP * 2 * DD];   // concat input / attention input (reused)
  __shared__ f16  sH [RP * DD];       // hidden
  __shared__ f16  sXI[RP * DD];       // x_ia (f16)
  __shared__ float sMask[RP];
  __shared__ float sAlpha[RP];
  __shared__ float sUW[DD];
  __shared__ float sVec[DD];

  const int* pg = pairs + (size_t)g * PP * 2;

  // gather item/rate embeddings -> sX = [emb_item | emb_rate], zero-pad rows 40..47
  for (int p = 0; p < RP; ++p) {
    int iid = 0, rid = 0;
    if (p < PP) { iid = pg[2 * p]; rid = pg[2 * p + 1]; }
    const float* ie = item_tab + (size_t)iid * DD;
    const float* re = rate_tab + (size_t)rid * DD;
    for (int c = lane; c < DD; c += 32) {
      sX[p * 2 * DD + c]      = (p < PP) ? (f16)ie[c] : (f16)0.0f;
      sX[p * 2 * DD + DD + c] = (p < PP) ? (f16)re[c] : (f16)0.0f;
    }
    if (lane == 0) sMask[p] = (p < PP && iid > 0) ? 1.0f : 0.0f;
  }

  // x_ia = (leaky(X @ gvW1 + b1)) @ gvW2 + b2
  wgemm<3>(sX, 2 * DD, 2 * DD, gvW1p, DD, gvb1, nullptr, nullptr, nullptr, nullptr, 1,
           sH, DD, 0, nullptr, 0, lane);
  wgemm<3>(sH, DD, DD, gvW2p, DD, gvb2, nullptr, nullptr, nullptr, nullptr, 0,
           sXI, DD, 0, nullptr, 0, lane);

  // attention input first half: x_ia @ wW + wb  -> sX cols [0,64)
  wgemm<3>(sXI, DD, DD, wWp, DD, wb, nullptr, nullptr, nullptr, nullptr, 0,
           sX, 2 * DD, 0, nullptr, 0, lane);
  // second half: (mask * user_emb) @ wW + wb  = mask ? uW : wb
  const float* ue = user_tab + (size_t)users[g] * DD;
  dense_vec(ue, wW, wb, sUW, DD, DD, lane, 0);
  for (int p = 0; p < RP; ++p)
    for (int c = lane; c < DD; c += 32) {
      float v = 0.0f;
      if (p < PP) v = (sMask[p] > 0.0f) ? sUW[c] : wb[c];
      sX[p * 2 * DD + DD + c] = (f16)v;
    }

  // attention MLP hidden
  wgemm<3>(sX, 2 * DD, 2 * DD, aW1p, DD, ab1, nullptr, nullptr, nullptr, nullptr, 1,
           sH, DD, 0, nullptr, 0, lane);
  // logits -> masked exp
  for (int p = lane; p < PP; p += 32) {
    float s = ab2[0];
    for (int c = 0; c < DD; ++c) s += (float)sH[p * DD + c] * aW2[c];
    sAlpha[p] = __expf(s) * sMask[p];
  }
  // normalize + weighted sum of x_ia
  float tot = 1e-10f;
  for (int p = 0; p < PP; ++p) tot += sAlpha[p];
  for (int c = lane; c < DD; c += 32) {
    float s = 0.0f;
    for (int p = 0; p < PP; ++p) s += sAlpha[p] * (float)sXI[p * DD + c];
    sVec[c] = s / tot;
  }
  // aggregation: relu(vec @ agW + agb)
  dense_vec(sVec, agW, agb, out + (size_t)g * DD, DD, DD, lane, 2);
}

// ---------------- Kernel 3: social-space attention over h_oI ----------------
__global__ __launch_bounds__(32) void k_social(
    const int* __restrict__ uids, const int* __restrict__ u_item_pad,
    const int* __restrict__ u_user_pad,
    const float* __restrict__ user_tab, const float* __restrict__ hoI, // [B*Q, DD]
    const f16* __restrict__ w5Wp, const float* __restrict__ w5W, const float* __restrict__ w5b,
    const f16* __restrict__ a5W1p, const float* __restrict__ a5b1,
    const float* __restrict__ a5W2, const float* __restrict__ a5b2,
    const float* __restrict__ ag5W, const float* __restrict__ ag5b,
    float* __restrict__ out)            // [B, DD]
{
  const int b = blockIdx.x;
  const int lane = threadIdx.x;
  __shared__ f16  sBin[RP * 2 * DD];
  __shared__ f16  sHo [RP * DD];
  __shared__ f16  sH  [RP * DD];
  __shared__ float sMaskU[RP];
  __shared__ float sBeta[RP];
  __shared__ float sUW[DD];
  __shared__ float sVec[DD];

  const float* ho = hoI + (size_t)b * PP * DD;
  for (int q = 0; q < RP; ++q)
    for (int c = lane; c < DD; c += 32)
      sHo[q * DD + c] = (q < PP) ? (f16)ho[(size_t)q * DD + c] : (f16)0.0f;
  for (int q = lane; q < PP; q += 32)
    sMaskU[q] = (u_item_pad[(size_t)b * PP * 2 + 2 * q] > 0) ? 1.0f : 0.0f;

  // b_in first half: h_oI @ w5 + b5
  wgemm<3>(sHo, DD, DD, w5Wp, DD, w5b, nullptr, nullptr, nullptr, nullptr, 0,
           sBin, 2 * DD, 0, nullptr, 0, lane);
  // second half: p_i @ w5 + b5 (p_i = mask_u * user_emb, broadcast)
  const float* ue = user_tab + (size_t)uids[b] * DD;
  dense_vec(ue, w5W, w5b, sUW, DD, DD, lane, 0);
  for (int q = 0; q < RP; ++q)
    for (int c = lane; c < DD; c += 32) {
      float v = 0.0f;
      if (q < PP) v = (sMaskU[q] > 0.0f) ? sUW[c] : w5b[c];
      sBin[q * 2 * DD + DD + c] = (f16)v;
    }

  wgemm<3>(sBin, 2 * DD, 2 * DD, a5W1p, DD, a5b1, nullptr, nullptr, nullptr, nullptr, 1,
           sH, DD, 0, nullptr, 0, lane);
  for (int q = lane; q < PP; q += 32) {
    float s = a5b2[0];
    for (int c = 0; c < DD; ++c) s += (float)sH[q * DD + c] * a5W2[c];
    float msk = (u_user_pad[(size_t)b * PP + q] > 0) ? 1.0f : 0.0f;
    sBeta[q] = __expf(s) * msk;
  }
  float tot = 1e-10f;
  for (int q = 0; q < PP; ++q) tot += sBeta[q];
  for (int c = lane; c < DD; c += 32) {
    float s = 0.0f;
    for (int q = 0; q < PP; ++q) s += sBeta[q] * ho[(size_t)q * DD + c];
    sVec[c] = s / tot;
  }
  dense_vec(sVec, ag5W, ag5b, out + (size_t)b * DD, DD, DD, lane, 2);
}

// ---------------- Kernel 4: combine MLP with BatchNorm (eval) ----------------
__global__ __launch_bounds__(32) void k_combine(
    const float* __restrict__ hiI, const float* __restrict__ hiS,
    const f16* __restrict__ c1Wp, const float* __restrict__ c1b,
    const float* __restrict__ bn1g, const float* __restrict__ bn1b,
    const float* __restrict__ bn1m, const float* __restrict__ bn1v,
    const f16* __restrict__ c2Wp, const float* __restrict__ c2b,
    const float* __restrict__ bn2g, const float* __restrict__ bn2b,
    const float* __restrict__ bn2m, const float* __restrict__ bn2v,
    const f16* __restrict__ c3Wp, const float* __restrict__ c3b,
    float* __restrict__ out)            // [B, DD]
{
  const int lane = threadIdx.x;
  const int r0 = blockIdx.x * 16;
  __shared__ f16 sA [16 * 3 * DD];    // [hiI*hiS | hiI | hiS]
  __shared__ f16 sB1[16 * 2 * DD];
  __shared__ f16 sB2[16 * DD];

  for (int r = 0; r < 16; ++r) {
    const float* a = hiI + (size_t)(r0 + r) * DD;
    const float* s = hiS + (size_t)(r0 + r) * DD;
    for (int c = lane; c < DD; c += 32) {
      float av = a[c], sv = s[c];
      sA[r * 3 * DD + c]          = (f16)(av * sv);
      sA[r * 3 * DD + DD + c]     = (f16)av;
      sA[r * 3 * DD + 2 * DD + c] = (f16)sv;
    }
  }
  wgemm<1>(sA, 3 * DD, 3 * DD, c1Wp, 2 * DD, c1b, bn1g, bn1b, bn1m, bn1v, 2,
           sB1, 2 * DD, 0, nullptr, 0, lane);
  wgemm<1>(sB1, 2 * DD, 2 * DD, c2Wp, DD, c2b, bn2g, bn2b, bn2m, bn2v, 2,
           sB2, DD, 0, nullptr, 0, lane);
  wgemm<1>(sB2, DD, DD, c3Wp, DD, c3b, nullptr, nullptr, nullptr, nullptr, 0,
           nullptr, 0, 0, out + (size_t)r0 * DD, DD, lane);
}

// ---------------- launch ----------------
extern "C" void kernel_launch(void* const* d_in, const int* in_sizes, int n_in,
                              void* d_out, int out_size, void* d_ws, size_t ws_size,
                              hipStream_t stream) {
  (void)in_sizes; (void)n_in; (void)out_size; (void)ws_size;
  const float* user_tab = (const float*)d_in[0];
  const float* item_tab = (const float*)d_in[1];
  const float* rate_tab = (const float*)d_in[2];
  const float* w1W = (const float*)d_in[3];  const float* w1b = (const float*)d_in[4];
  const float* w4W = (const float*)d_in[5];  const float* w4b = (const float*)d_in[6];
  const float* w5W = (const float*)d_in[7];  const float* w5b = (const float*)d_in[8];
  const float* gvW1 = (const float*)d_in[9];  const float* gvb1 = (const float*)d_in[10];
  const float* gvW2 = (const float*)d_in[11]; const float* gvb2 = (const float*)d_in[12];
  const float* a1W1 = (const float*)d_in[13]; const float* a1b1 = (const float*)d_in[14];
  const float* a1W2 = (const float*)d_in[15]; const float* a1b2 = (const float*)d_in[16];
  const float* ag1W = (const float*)d_in[17]; const float* ag1b = (const float*)d_in[18];
  const float* a4W1 = (const float*)d_in[19]; const float* a4b1 = (const float*)d_in[20];
  const float* a4W2 = (const float*)d_in[21]; const float* a4b2 = (const float*)d_in[22];
  const float* ag4W = (const float*)d_in[23]; const float* ag4b = (const float*)d_in[24];
  const float* a5W1 = (const float*)d_in[25]; const float* a5b1 = (const float*)d_in[26];
  const float* a5W2 = (const float*)d_in[27]; const float* a5b2 = (const float*)d_in[28];
  const float* ag5W = (const float*)d_in[29]; const float* ag5b = (const float*)d_in[30];
  const float* c1W = (const float*)d_in[31];  const float* c1b = (const float*)d_in[32];
  const float* bn1g = (const float*)d_in[33]; const float* bn1b = (const float*)d_in[34];
  const float* bn1m = (const float*)d_in[35]; const float* bn1v = (const float*)d_in[36];
  const float* c2W = (const float*)d_in[37];  const float* c2b = (const float*)d_in[38];
  const float* bn2g = (const float*)d_in[39]; const float* bn2b = (const float*)d_in[40];
  const float* bn2m = (const float*)d_in[41]; const float* bn2v = (const float*)d_in[42];
  const float* c3W = (const float*)d_in[43];  const float* c3b = (const float*)d_in[44];
  const int* uids            = (const int*)d_in[45];
  const int* u_item_pad      = (const int*)d_in[46];  // [B,P,2]
  const int* u_user_pad      = (const int*)d_in[47];  // [B,Q]
  const int* u_user_item_pad = (const int*)d_in[48];  // [B,Q,P2,2]

  float* ws  = (float*)d_ws;
  float* hoI = ws;                                  // [B*Q, DD]
  float* hiI = hoI + (size_t)BB * PP * DD;          // [B, DD]
  float* hiS = hiI + (size_t)BB * DD;               // [B, DD]
  f16*   pk  = (f16*)(hiS + (size_t)BB * DD);       // packed f16 weights (32B aligned)

  f16* gvW1p = pk;               // 128x64 -> 8192 halves
  f16* gvW2p = gvW1p + 8192;     // 64x64  -> 4096
  f16* w1p   = gvW2p + 4096;
  f16* w4p   = w1p   + 4096;
  f16* w5p   = w4p   + 4096;
  f16* a1W1p = w5p   + 4096;     // 128x64 -> 8192
  f16* a4W1p = a1W1p + 8192;
  f16* a5W1p = a4W1p + 8192;
  f16* c1Wp  = a5W1p + 8192;     // 192x128 -> 24576
  f16* c2Wp  = c1Wp  + 24576;    // 128x64 -> 8192
  f16* c3Wp  = c2Wp  + 8192;     // 64x64 -> 4096

  // ---- pack weights into WMMA B-fragment order (f16), once per launch ----
  k_pack_w<<<(128/32)*(64/16),  32, 0, stream>>>(gvW1, 128, 64,  gvW1p);
  k_pack_w<<<(64/32)*(64/16),   32, 0, stream>>>(gvW2, 64,  64,  gvW2p);
  k_pack_w<<<(64/32)*(64/16),   32, 0, stream>>>(w1W,  64,  64,  w1p);
  k_pack_w<<<(64/32)*(64/16),   32, 0, stream>>>(w4W,  64,  64,  w4p);
  k_pack_w<<<(64/32)*(64/16),   32, 0, stream>>>(w5W,  64,  64,  w5p);
  k_pack_w<<<(128/32)*(64/16),  32, 0, stream>>>(a1W1, 128, 64,  a1W1p);
  k_pack_w<<<(128/32)*(64/16),  32, 0, stream>>>(a4W1, 128, 64,  a4W1p);
  k_pack_w<<<(128/32)*(64/16),  32, 0, stream>>>(a5W1, 128, 64,  a5W1p);
  k_pack_w<<<(192/32)*(128/16), 32, 0, stream>>>(c1W,  192, 128, c1Wp);
  k_pack_w<<<(128/32)*(64/16),  32, 0, stream>>>(c2W,  128, 64,  c2Wp);
  k_pack_w<<<(64/32)*(64/16),   32, 0, stream>>>(c3W,  64,  64,  c3Wp);

  // first-order item aggregation -> h_iI
  k_attn_agg<<<BB, 32, 0, stream>>>(
      u_item_pad, uids, item_tab, rate_tab, user_tab,
      gvW1p, gvb1, gvW2p, gvb2, w1p, w1W, w1b, a1W1p, a1b1, a1W2, a1b2, ag1W, ag1b, hiI);
  // second-order: neighbors' item aggregation -> h_oI (dominant: 20480 waves)
  k_attn_agg<<<BB * PP, 32, 0, stream>>>(
      u_user_item_pad, u_user_pad, item_tab, rate_tab, user_tab,
      gvW1p, gvb1, gvW2p, gvb2, w4p, w4W, w4b, a4W1p, a4b1, a4W2, a4b2, ag4W, ag4b, hoI);
  // social aggregation -> h_iS
  k_social<<<BB, 32, 0, stream>>>(
      uids, u_item_pad, u_user_pad, user_tab, hoI,
      w5p, w5W, w5b, a5W1p, a5b1, a5W2, a5b2, ag5W, ag5b, hiS);
  // combine MLP -> out
  k_combine<<<BB / 16, 32, 0, stream>>>(
      hiI, hiS, c1Wp, c1b, bn1g, bn1b, bn1m, bn1v,
      c2Wp, c2b, bn2g, bn2b, bn2m, bn2v, c3Wp, c3b, (float*)d_out);
}
